// FeedForwardQuantum_65481071396095
// MI455X (gfx1250) — matrix operand order repaired and checked by reading, not compiled, not run on hardware
//
#include <hip/hip_runtime.h>

// CDNA5 / gfx1250 fused "quantum FFN":
//   q = cos(2*x[...,:8] + theta); h = relu(q @ W1^T); out = h @ W2^T
// GEMM2 (32768x2048x512, 68.7 GFLOP): v_wmma_f32_16x16x32_f16, f16 in / f32 acc.
// h lives only in LDS. W2 pre-converted to f16 in d_ws once per launch; the
// main kernel streams it into LDS with GLOBAL_LOAD_ASYNC_TO_LDS_B128
// (ASYNCcnt path) when the builtin is available.

typedef __attribute__((ext_vector_type(16))) _Float16 v16h;
typedef __attribute__((ext_vector_type(8)))  float    v8f;
typedef __attribute__((ext_vector_type(4)))  int      v4i;

#define R_TOTAL 32768   // BATCH * SEQ
#define EMBED   512
#define FFN     2048
#define NQ      8

#define MT 64           // M rows per workgroup
#define NT 256          // N cols per workgroup
#define KC 32           // K chunk (one WMMA K-step)

#define H_STRIDE 40     // halves per sH row (80 B, bank-spread, 16B aligned)
#define B_STRIDE 40     // halves per sB row

#if defined(__AMDGCN__) && __has_builtin(__builtin_amdgcn_global_load_async_to_lds_b128) && \
    __has_builtin(__builtin_amdgcn_s_wait_asynccnt)
#define FFQ_USE_ASYNC 1
typedef __attribute__((address_space(1))) v4i* gv4i_p;   // global int4*
typedef __attribute__((address_space(3))) v4i* lv4i_p;   // LDS int4*
#else
#define FFQ_USE_ASYNC 0
#endif

// ---------------------------------------------------------------------------
// Prep: W2 f32 -> f16 (once per launch; W2 is 4MB, L2-resident afterwards)
// ---------------------------------------------------------------------------
__global__ __launch_bounds__(256)
void w2_to_f16(const float* __restrict__ W2, unsigned short* __restrict__ W2h)
{
    const int idx = blockIdx.x * 256 + threadIdx.x;       // float4 index
    const float4 v = ((const float4*)W2)[idx];
    __align__(8) _Float16 h4[4] = { (_Float16)v.x, (_Float16)v.y,
                                    (_Float16)v.z, (_Float16)v.w };
    ((uint2*)W2h)[idx] = *(const uint2*)h4;
}

// ---------------------------------------------------------------------------
// Main fused kernel, B operand streamed from pre-converted f16 W2
// ---------------------------------------------------------------------------
__global__ __launch_bounds__(256)
void ffq_fused_wmma_f16w(const float* __restrict__ x,
                         const float* __restrict__ theta,
                         const float* __restrict__ W1,          // [FFN, 8] f32
                         const unsigned short* __restrict__ W2h,// [EMBED, FFN] f16
                         float* __restrict__ out)               // [R, EMBED] f32
{
    __shared__ __align__(16) float    sQ[MT][NQ];
    __shared__ __align__(16) _Float16 sH[MT * H_STRIDE];   // A staging (f16)
    __shared__ __align__(16) _Float16 sB[NT * B_STRIDE];   // B staging (f16)

    const int tid  = threadIdx.x;
    const int wave = tid >> 5;
    const int lane = tid & 31;
    const int mw   = wave >> 1;     // 0..3 : 16-row strip
    const int nw   = wave & 1;      // 0..1 : 128-col strip
    const int lane15 = lane & 15;
    const bool loHalf = (lane < 16);

    const int mBase = blockIdx.x * MT;
    const int eBase = blockIdx.y * NT;

    // ---- Phase 0: q = cos(2x + theta) for this block's 64 rows ----
    for (int i = tid; i < MT * NQ; i += 256) {
        const int m = i >> 3, j = i & 7;
        const float xv = x[(size_t)(mBase + m) * EMBED + j];
        sQ[m][j] = __builtin_cosf(2.0f * xv + theta[j]);
    }
    __syncthreads();

    // A fragment LDS offsets (16-bit A 16x32 layout):
    //  lanes 0-15 : M=lane,    K = {0..7, 16..23}
    //  lanes 16-31: M=lane-16, K = {8..15, 24..31}
    const int aRow  = mw * 16 + lane15;
    const int aOff0 = aRow * H_STRIDE + (loHalf ? 0 : 8);
    const int aOff1 = aOff0 + 16;

    v8f acc[8];
    {
        v8f zero = {};
        #pragma unroll
        for (int t = 0; t < 8; ++t) acc[t] = zero;
    }

    for (int kc = 0; kc < FFN; kc += KC) {
        // ---- stage W2 chunk -> sB[e][k] (pure f16 copy, 4 x 16B per thread) ----
        #pragma unroll
        for (int it = 0; it < 4; ++it) {
            const int idx = it * 256 + tid;     // 0..1023 16B tiles
            const int e   = idx >> 2;           // 0..255
            const int kq  = idx & 3;            // 8-half group within 32 k
            const unsigned short* gsrc =
                &W2h[(size_t)(eBase + e) * FFN + kc + kq * 8];
            _Float16* ldst = &sB[e * B_STRIDE + kq * 8];
#if FFQ_USE_ASYNC
            __builtin_amdgcn_global_load_async_to_lds_b128(
                (gv4i_p)gsrc, (lv4i_p)ldst, /*offset=*/0, /*cpol=*/0);
#else
            *(uint4*)ldst = *(const uint4*)gsrc;
#endif
        }

        // ---- stage h chunk: sH[m][k] = relu(dot8(q[m], W1 row kc+k)) as f16 ----
        {
            const int m  = tid >> 2;         // 0..63
            const int k0 = (tid & 3) * 8;    // 0,8,16,24
            float q0[NQ];
            #pragma unroll
            for (int j = 0; j < NQ; ++j) q0[j] = sQ[m][j];

            __align__(16) _Float16 hv[8];
            #pragma unroll
            for (int i = 0; i < 8; ++i) {
                const float* w = &W1[(size_t)(kc + k0 + i) * NQ];
                float s = 0.0f;
                #pragma unroll
                for (int j = 0; j < NQ; ++j) s = fmaf(q0[j], w[j], s);
                hv[i] = (_Float16)fmaxf(s, 0.0f);
            }
            *(uint4*)&sH[m * H_STRIDE + k0] = *(const uint4*)hv;
        }

#if FFQ_USE_ASYNC
        __builtin_amdgcn_s_wait_asynccnt(0);   // this wave's async B copies done
#endif
        __syncthreads();                       // all waves' sH/sB visible

        // ---- WMMA: one A fragment, 8 B tiles (16 N each) ----
        v16h a;
        *((uint4*)&a)     = *(const uint4*)&sH[aOff0];
        *((uint4*)&a + 1) = *(const uint4*)&sH[aOff1];

        #pragma unroll
        for (int t = 0; t < 8; ++t) {
            // B 32x16 f16 layout: lanes 0-15 N=lane K={0..7,16..23} ; lanes 16-31 K={8..15,24..31}
            const int bRow = nw * 128 + t * 16 + lane15;
            const int bOff = bRow * B_STRIDE + (loHalf ? 0 : 8);
            v16h b;
            *((uint4*)&b)     = *(const uint4*)&sB[bOff];
            *((uint4*)&b + 1) = *(const uint4*)&sB[bOff + 16];
            acc[t] = __builtin_amdgcn_wmma_f32_16x16x32_f16(
                         false, a, false, b, (short)0, acc[t], false, false);
        }
        __syncthreads();
    }

    // ---- store D: VGPR r -> M = r (lanes 0-15) / 8+r (lanes 16-31), N = lane15
    const int mRow0 = mBase + mw * 16 + (loHalf ? 0 : 8);
    const int col0  = eBase + nw * 128 + lane15;
    #pragma unroll
    for (int t = 0; t < 8; ++t) {
        #pragma unroll
        for (int r = 0; r < 8; ++r) {
            out[(size_t)(mRow0 + r) * EMBED + (col0 + t * 16)] = acc[t][r];
        }
    }
}

// ---------------------------------------------------------------------------
// Fallback (round-1 kernel): fully fused, converts W2 on the fly. Used only
// if d_ws cannot hold the 2MB f16 copy of W2.
// ---------------------------------------------------------------------------
__global__ __launch_bounds__(256)
void ffq_fused_wmma(const float* __restrict__ x,
                    const float* __restrict__ theta,
                    const float* __restrict__ W1,
                    const float* __restrict__ W2,
                    float* __restrict__ out)
{
    __shared__ __align__(16) float    sQ[MT][NQ];
    __shared__ __align__(16) _Float16 sH[MT * H_STRIDE];
    __shared__ __align__(16) _Float16 sB[NT * B_STRIDE];

    const int tid  = threadIdx.x;
    const int wave = tid >> 5;
    const int lane = tid & 31;
    const int mw   = wave >> 1;
    const int nw   = wave & 1;
    const int lane15 = lane & 15;
    const bool loHalf = (lane < 16);

    const int mBase = blockIdx.x * MT;
    const int eBase = blockIdx.y * NT;

    for (int i = tid; i < MT * NQ; i += 256) {
        const int m = i >> 3, j = i & 7;
        const float xv = x[(size_t)(mBase + m) * EMBED + j];
        sQ[m][j] = __builtin_cosf(2.0f * xv + theta[j]);
    }
    __syncthreads();

    const int aRow  = mw * 16 + lane15;
    const int aOff0 = aRow * H_STRIDE + (loHalf ? 0 : 8);
    const int aOff1 = aOff0 + 16;

    v8f acc[8];
    {
        v8f zero = {};
        #pragma unroll
        for (int t = 0; t < 8; ++t) acc[t] = zero;
    }

    for (int kc = 0; kc < FFN; kc += KC) {
        {
            const int m  = tid >> 2;
            const int k0 = (tid & 3) * 8;
            float q0[NQ];
            #pragma unroll
            for (int j = 0; j < NQ; ++j) q0[j] = sQ[m][j];

            __align__(16) _Float16 hv[8];
            #pragma unroll
            for (int i = 0; i < 8; ++i) {
                const float* w = &W1[(size_t)(kc + k0 + i) * NQ];
                float s = 0.0f;
                #pragma unroll
                for (int j = 0; j < NQ; ++j) s = fmaf(q0[j], w[j], s);
                hv[i] = (_Float16)fmaxf(s, 0.0f);
            }
            *(uint4*)&sH[m * H_STRIDE + k0] = *(const uint4*)hv;
        }
        #pragma unroll
        for (int it = 0; it < 8; ++it) {
            const int idx = it * 256 + tid;
            const int e   = idx >> 3;
            const int kq  = idx & 7;
            const float4 wv =
                *(const float4*)&W2[(size_t)(eBase + e) * FFN + kc + kq * 4];
            __align__(8) _Float16 h4[4] = { (_Float16)wv.x, (_Float16)wv.y,
                                            (_Float16)wv.z, (_Float16)wv.w };
            *(uint2*)&sB[e * B_STRIDE + kq * 4] = *(const uint2*)h4;
        }
        __syncthreads();

        v16h a;
        *((uint4*)&a)     = *(const uint4*)&sH[aOff0];
        *((uint4*)&a + 1) = *(const uint4*)&sH[aOff1];

        #pragma unroll
        for (int t = 0; t < 8; ++t) {
            const int bRow = nw * 128 + t * 16 + lane15;
            const int bOff = bRow * B_STRIDE + (loHalf ? 0 : 8);
            v16h b;
            *((uint4*)&b)     = *(const uint4*)&sB[bOff];
            *((uint4*)&b + 1) = *(const uint4*)&sB[bOff + 16];
            acc[t] = __builtin_amdgcn_wmma_f32_16x16x32_f16(
                         false, a, false, b, (short)0, acc[t], false, false);
        }
        __syncthreads();
    }

    const int mRow0 = mBase + mw * 16 + (loHalf ? 0 : 8);
    const int col0  = eBase + nw * 128 + lane15;
    #pragma unroll
    for (int t = 0; t < 8; ++t) {
        #pragma unroll
        for (int r = 0; r < 8; ++r) {
            out[(size_t)(mRow0 + r) * EMBED + (col0 + t * 16)] = acc[t][r];
        }
    }
}

extern "C" void kernel_launch(void* const* d_in, const int* in_sizes, int n_in,
                              void* d_out, int out_size, void* d_ws, size_t ws_size,
                              hipStream_t stream) {
    const float* x     = (const float*)d_in[0];   // [16, 2048, 512] f32
    const float* theta = (const float*)d_in[1];   // [8] f32
    const float* W1    = (const float*)d_in[2];   // [2048, 8] f32
    const float* W2    = (const float*)d_in[3];   // [512, 2048] f32
    float* out = (float*)d_out;                   // [16, 2048, 512] f32

    const size_t w2h_bytes = (size_t)EMBED * FFN * sizeof(unsigned short); // 2 MB
    dim3 grid(R_TOTAL / MT, EMBED / NT);          // (512, 2)

    if (ws_size >= w2h_bytes) {
        unsigned short* W2h = (unsigned short*)d_ws;
        // 512*2048 floats = 262144 float4 tiles -> 1024 blocks x 256 threads
        w2_to_f16<<<(EMBED * FFN) / (256 * 4), 256, 0, stream>>>(W2, W2h);
        ffq_fused_wmma_f16w<<<grid, 256, 0, stream>>>(x, theta, W1, W2h, out);
    } else {
        ffq_fused_wmma<<<grid, 256, 0, stream>>>(x, theta, W1, W2, out);
    }
}
